// VisualAdapter_6579889898085
// MI455X (gfx1250) — matrix operand order repaired and checked by reading, not compiled, not run on hardware
//
#include <hip/hip_runtime.h>
#include <hip/hip_bf16.h>

// ---------------- problem constants ----------------
constexpr int kB  = 64;
constexpr int kC  = 768;
constexpr int kN  = 784;
constexpr int kNV = 784;
constexpr int kT  = 32;
constexpr int kDS = 48;     // down-sample channels
constexpr int kCg = 384;    // C / G
constexpr int kDSg = 24;    // DS / G
constexpr float kEPS = 1e-5f;

constexpr int NT = 112;     // n-tile (7 tiles of 112 cover 784)
constexpr int NTILES = 7;

// ---------------- workspace layout (floats) ----------------
constexpr size_t WS_REP   = 0;                                   // [B][T][C]
constexpr size_t WS_Y1    = WS_REP + (size_t)kB * kT * kC;       // [B][DS][N]
constexpr size_t WS_STATS = WS_Y1 + (size_t)kB * kDS * kN;
constexpr size_t WS_SUM1  = WS_STATS;            // 48
constexpr size_t WS_SQ1   = WS_SUM1 + 48;        // 48
constexpr size_t WS_SUM2  = WS_SQ1 + 48;         // 768
constexpr size_t WS_SQ2   = WS_SUM2 + 768;       // 768
constexpr int    N_ZERO   = 48 + 48 + 768 + 768; // 1632
constexpr size_t WS_SC1   = WS_SQ2 + 768;
constexpr size_t WS_SH1   = WS_SC1 + 48;
constexpr size_t WS_SC2   = WS_SH1 + 48;
constexpr size_t WS_SH2   = WS_SC2 + 768;

// ---------------- WMMA plumbing ----------------
typedef __bf16 bf16;
typedef __attribute__((ext_vector_type(16))) __bf16 v16bf;
typedef __attribute__((ext_vector_type(8)))  float  v8f;
typedef __attribute__((ext_vector_type(4)))  unsigned int v4u;
typedef __attribute__((ext_vector_type(4)))  int    v4i;
typedef __attribute__((ext_vector_type(8)))  int    v8i;

__device__ __forceinline__ v8f wmma_bf16(v16bf a, v16bf b, v8f c) {
  // D = A(16x32 bf16) * B(32x16 bf16) + C(16x16 f32)
  return __builtin_amdgcn_wmma_f32_16x16x32_bf16(false, a, false, b, (short)0, c,
                                                 false, false);
}

// A-matrix 16x32 bf16 layout: lane l holds row M=l&15;
// lanes<16 see K=0..15, lanes>=16 see K=16..31 (matches B).
__device__ __forceinline__ int a_kmap(int e, int kb) {
  return 2 * kb + e;
}

// A[m][k] = p[m*ld + k]   (row-major source; contiguous in k -> ds_load_b128)
__device__ __forceinline__ v16bf load_a(const bf16* p, int ld) {
  int l = threadIdx.x & 31;
  int m = l & 15, kb = (l >> 4) << 3;
  v16bf a;
#pragma unroll
  for (int e = 0; e < 16; ++e) a[e] = p[m * ld + a_kmap(e, kb)];
  return a;
}

// A[m][k] = (bf16)p[m*ld + k]   (fp32 source, convert)
__device__ __forceinline__ v16bf load_a_f32(const float* p, int ld) {
  int l = threadIdx.x & 31;
  int m = l & 15, kb = (l >> 4) << 3;
  v16bf a;
#pragma unroll
  for (int e = 0; e < 16; ++e) a[e] = (bf16)p[m * ld + a_kmap(e, kb)];
  return a;
}

// B-matrix 32x16: lane l holds col N=l&15; lanes<16: K=0..15, lanes>=16: K=16..31.
// B[k][n] = p[k*ld + n]
__device__ __forceinline__ v16bf load_b(const bf16* p, int ld) {
  int l = threadIdx.x & 31;
  int n = l & 15, kb = (l >> 4) << 4;
  v16bf b;
#pragma unroll
  for (int e = 0; e < 16; ++e) b[e] = p[(kb + e) * ld + n];
  return b;
}

// B[k][n] = p[n*ld + k]   (transposed source; contiguous in k)
__device__ __forceinline__ v16bf load_bT(const bf16* p, int ld) {
  int l = threadIdx.x & 31;
  int n = l & 15, kb = (l >> 4) << 4;
  v16bf b;
#pragma unroll
  for (int e = 0; e < 16; ++e) b[e] = p[n * ld + kb + e];
  return b;
}

// C/D 16x16 f32: VGPR r -> lanes 0-15 (M=r,N=lane), lanes 16-31 (M=r+8,N=lane-16)
__device__ __forceinline__ void store_c(float* p, int ld, v8f c) {
  int l = threadIdx.x & 31;
  int n = l & 15, mo = (l >> 4) << 3;
#pragma unroll
  for (int r = 0; r < 8; ++r) p[(r + mo) * ld + n] = c[r];
}

// ---------------- Tensor Data Mover: 1D/2D f32 tile -> LDS ----------------
// D# per CDNA5 ISA ch.8: group0 {count, lds_addr, global_addr, type=2},
// group1 {data_size=4B, tensor dims, tile dims, dim0 stride}. Issued by one
// wave (TDM ignores EXEC); completion via s_wait_tensorcnt.
__device__ __forceinline__ void tdm_load_f32(unsigned lds_off, const float* gptr,
                                             unsigned tile_d0, unsigned tile_d1,
                                             unsigned stride_d0) {
  unsigned long long ga = (unsigned long long)(size_t)gptr;
  v4u g0;
  g0[0] = 1u;                                            // count=1 (user D#)
  g0[1] = lds_off;                                       // lds_addr (bytes)
  g0[2] = (unsigned)(ga & 0xFFFFFFFFu);                  // global_addr[31:0]
  g0[3] = (unsigned)((ga >> 32) & 0x01FFFFFFu) | (2u << 30); // addr[56:32]|type=2
  v8i g1;
  g1[0] = (int)(2u << 16);                               // data_size = 4B
  g1[1] = (int)((tile_d0 & 0xFFFFu) << 16);              // tensor_dim0[15:0]
  g1[2] = (int)(((tile_d0 >> 16) & 0xFFFFu) |            // tensor_dim0[31:16]
                ((tile_d1 & 0xFFFFu) << 16));            // tensor_dim1[15:0]
  g1[3] = (int)(((tile_d1 >> 16) & 0xFFFFu) |            // tensor_dim1[31:16]
                ((tile_d0 & 0xFFFFu) << 16));            // tile_dim0
  g1[4] = (int)(tile_d1 & 0xFFFFu);                      // tile_dim1 (0 => 1D)
  g1[5] = (int)stride_d0;                                // tensor_dim0_stride lo
  g1[6] = 0;
  g1[7] = 0;
  v4i z4 = {0, 0, 0, 0};
#if defined(__clang_major__) && (__clang_major__ >= 23)
  v8i z8 = {0, 0, 0, 0, 0, 0, 0, 0};
  __builtin_amdgcn_tensor_load_to_lds(g0, g1, z4, z4, z8, 0);
#else
  __builtin_amdgcn_tensor_load_to_lds(g0, g1, z4, z4, 0);
#endif
}

// ---------------- kernel 0: zero BN accumulators ----------------
__global__ void k0_init(float* p, int n) {
  int i = blockIdx.x * blockDim.x + threadIdx.x;
  if (i < n) p[i] = 0.f;
}

// ---------------- kernel 1: rep = tok + softmax(tok·vt)·vt^T  (per batch) ---
// grid: 64 blocks, 256 threads, LDS: tokens bf16 (48KB) + att f32 32x800 (100KB)
__global__ void k1_rep(const float* __restrict__ vt, const float* __restrict__ tok,
                       float* __restrict__ rep) {
  extern __shared__ char smem[];
  bf16*  tokA = (bf16*)smem;                    // [32][768] bf16
  float* att  = (float*)(smem + kT * kC * 2);   // [32][800] f32 (K padded)
  const int b = blockIdx.x, tid = threadIdx.x;
  const float* vtb = vt + (size_t)b * kC * kNV;

  for (int i = tid; i < kT * kC; i += 256) tokA[i] = (bf16)tok[i];
  __syncthreads();

  const int wave = tid >> 5, lane = tid & 31;
  const int ln = lane & 15, hi = lane >> 4;

  // ---- stage 1: S[t][n] = sum_c tok[t][c] * vt[c][n] ----
  for (int j = wave; j < kNV / 16; j += 8) {
    const int n0 = j * 16;
    v8f acc0 = {}, acc1 = {};
    for (int k0 = 0; k0 < kC; k0 += 32) {
      v16bf a0 = load_a(tokA + k0, kC);
      v16bf a1 = load_a(tokA + 16 * kC + k0, kC);
      v16bf bf;
      const int kb = hi * 16;
      if (k0 + 32 < kC)
        __builtin_prefetch(&vtb[(size_t)(k0 + 32 + kb) * kNV + n0 + ln], 0, 1);
#pragma unroll
      for (int e = 0; e < 16; ++e)
        bf[e] = (bf16)vtb[(size_t)(k0 + kb + e) * kNV + n0 + ln];
      acc0 = wmma_bf16(a0, bf, acc0);
      acc1 = wmma_bf16(a1, bf, acc1);
    }
    store_c(att + n0, 800, acc0);
    store_c(att + 16 * 800 + n0, 800, acc1);
  }
  __syncthreads();

  // ---- softmax over n (one wave per row) ----
  for (int t = wave; t < kT; t += 8) {
    float* row = att + t * 800;
    float mx = -1e30f;
    for (int n = lane; n < kNV; n += 32) mx = fmaxf(mx, row[n]);
    for (int o = 16; o > 0; o >>= 1) mx = fmaxf(mx, __shfl_xor(mx, o, 32));
    float s = 0.f;
    for (int n = lane; n < kNV; n += 32) { float e = __expf(row[n] - mx); row[n] = e; s += e; }
    for (int o = 16; o > 0; o >>= 1) s += __shfl_xor(s, o, 32);
    const float inv = 1.f / s;
    for (int n = lane; n < kNV; n += 32) row[n] *= inv;
    for (int n = kNV + lane; n < 800; n += 32) row[n] = 0.f;  // K pad
  }
  __syncthreads();

  // ---- stage 2: rep[t][c] = tok[t][c] + sum_n att[t][n] * vt[c][n] ----
  float* repb = rep + (size_t)b * kT * kC;
  for (int j = wave; j < kC / 16; j += 8) {
    const int c0 = j * 16;
    v8f acc0 = {}, acc1 = {};
    for (int k0 = 0; k0 < 800; k0 += 32) {
      v16bf a0 = load_a_f32(att + k0, 800);
      v16bf a1 = load_a_f32(att + 16 * 800 + k0, 800);
      v16bf bf;
      const int kb = hi * 16;
#pragma unroll
      for (int e = 0; e < 16; ++e) {
        const int k = k0 + kb + e;
        bf[e] = (k < kNV) ? (bf16)vtb[(size_t)(c0 + ln) * kNV + k] : (bf16)0.f;
      }
      acc0 = wmma_bf16(a0, bf, acc0);
      acc1 = wmma_bf16(a1, bf, acc1);
    }
    const int mo = hi * 8;
#pragma unroll
    for (int r = 0; r < 8; ++r) {
      const int t = mo + r, c = c0 + ln;
      repb[t * kC + c]        = tok[t * kC + c] + acc0[r];
      repb[(16 + t) * kC + c] = tok[(16 + t) * kC + c] + acc1[r];
    }
  }
}

// ---------------- kernel 2: scores, softmax, x_res, h, down-conv, BN1 sums --
// grid: (7, 64), 256 threads. LDS ~306KB. rep staged via TDM (1-D descriptor).
constexpr unsigned K2_U     = 0;                       // 172032: xsb/hb union (+rep f32 stage)
constexpr unsigned K2_REPB  = 172032;                  // 49152 : repBf [32][768]
constexpr unsigned K2_REPT  = K2_REPB + 49152;         // 49152 : repT  [768][32]
constexpr unsigned K2_ATT   = K2_REPT + 49152;         // 14336 : att2  [112][32] f32
constexpr unsigned K2_ATTB  = K2_ATT + 14336;          // 7168  : att2b [112][32] bf16
constexpr unsigned K2_Y1T   = K2_ATTB + 7168;          // 21504 : y1t   [48][112] f32
constexpr unsigned K2_SMEM  = K2_Y1T + 21504;          // 313344 total

__global__ void k2_fused(const float* __restrict__ x, const float* __restrict__ rep,
                         const float* __restrict__ gate_av,
                         const float* __restrict__ w_down,
                         float* __restrict__ y1, float* __restrict__ sum1,
                         float* __restrict__ sumsq1) {
  extern __shared__ char smem[];
  bf16*  U     = (bf16*)(smem + K2_U);
  bf16*  repBf = (bf16*)(smem + K2_REPB);
  bf16*  repT  = (bf16*)(smem + K2_REPT);
  float* att2  = (float*)(smem + K2_ATT);
  bf16*  att2b = (bf16*)(smem + K2_ATTB);
  float* y1t   = (float*)(smem + K2_Y1T);

  const int b = blockIdx.y, n0 = blockIdx.x * NT, tid = threadIdx.x;
  const int wave = tid >> 5, lane = tid & 31;
  const int ln = lane & 15, hi = lane >> 4;
  const float* xb = x + (size_t)b * kC * kN;

  // ---- TDM: stage rep[b] (32x768 f32, contiguous) into U region ----
  if (wave == 0) {
    tdm_load_f32(K2_U, rep + (size_t)b * kT * kC, kT * kC, 0, 0);
    __builtin_amdgcn_s_wait_tensorcnt(0);
  }
  __syncthreads();

  // convert staged f32 rep -> bf16 (row-major + transposed copies)
  {
    const float* repF = (const float*)U;
    for (int i = tid; i < kT * kC; i += 256) {
      const int t = i / kC, c = i % kC;
      const bf16 v = (bf16)repF[i];
      repBf[i] = v;
      repT[c * kT + t] = v;
    }
  }
  __syncthreads();   // all reads of repF done before U is overwritten

  for (int i = tid; i < kC * NT; i += 256) {       // xsb[nl][c] = x[c][n0+nl]
    const int c = i / NT, nl = i % NT;
    U[nl * kC + c] = (bf16)xb[(size_t)c * kN + n0 + nl];
  }
  __syncthreads();

  // ---- scores[nl][t] = sum_c xs[c][nl] * rep[t][c]  (M=112, N=32, K=768) ----
  for (int mt = wave; mt < NTILES; mt += 8) {
    const int m0 = mt * 16;
    v8f a0 = {}, a1 = {};
    for (int k0 = 0; k0 < kC; k0 += 32) {
      v16bf av = load_a(U + m0 * kC + k0, kC);
      v16bf b0 = load_bT(repBf + k0, kC);             // t = 0..15
      v16bf b1 = load_bT(repBf + 16 * kC + k0, kC);   // t = 16..31
      a0 = wmma_bf16(av, b0, a0);
      a1 = wmma_bf16(av, b1, a1);
    }
    store_c(att2 + m0 * 32, 32, a0);
    store_c(att2 + m0 * 32 + 16, 32, a1);
  }
  __syncthreads();

  // ---- softmax over t (length 32) per n row ----
  if (tid < NT) {
    float* row = att2 + tid * 32;
    float mx = -1e30f;
    for (int t = 0; t < kT; ++t) mx = fmaxf(mx, row[t]);
    float s = 0.f;
    for (int t = 0; t < kT; ++t) s += __expf(row[t] - mx);
    const float inv = 1.f / s;
    for (int t = 0; t < kT; ++t) att2b[tid * 32 + t] = (bf16)(__expf(row[t] - mx) * inv);
  }
  __syncthreads();

  // ---- x_res[c][nl] = sum_t rep[t][c]*att[nl][t]; h = xs + g*x_res (K=32) ----
  const float g = gate_av[0];
  for (int j = wave; j < (kC / 16) * NTILES; j += 8) {
    const int ct = j / NTILES, nt = j % NTILES;
    const int c0 = ct * 16, j0 = nt * 16;
    v16bf av = load_a(repT + c0 * kT, kT);            // A[m=c][k=t], contiguous
    v16bf bv = load_bT(att2b + j0 * 32, 32);
    v8f acc = {};
    acc = wmma_bf16(av, bv, acc);
    const int mo = hi * 8;
#pragma unroll
    for (int r = 0; r < 8; ++r) {
      const int c = c0 + mo + r, nl = j0 + ln;
      const float hv = xb[(size_t)c * kN + n0 + nl] + g * acc[r];
      U[c * NT + nl] = (bf16)hv;                      // hb[c][nl] (reuses xsb space)
    }
  }
  __syncthreads();

  // ---- grouped down-conv: y1[o][nl] = sum_c Wd[o][c] * h[c][nl] (masked A) --
  for (int j = wave; j < (kDS / 16) * NTILES; j += 8) {
    const int mt = j / NTILES, nt = j % NTILES;
    const int m0 = mt * 16, j0 = nt * 16;
    const int m = m0 + ln;            // this lane's output row
    const int ga = m / kDSg;          // its group
    v8f acc = {};
    for (int k0 = 0; k0 < kC; k0 += 32) {
      v16bf av;
      const int kb = hi * 8;
#pragma unroll
      for (int e = 0; e < 16; ++e) {
        const int k = k0 + a_kmap(e, kb);
        const int cg = k - ga * kCg;
        av[e] = (cg >= 0 && cg < kCg) ? (bf16)w_down[m * kCg + cg] : (bf16)0.f;
      }
      v16bf bv = load_b(U + k0 * NT + j0, NT);
      acc = wmma_bf16(av, bv, acc);
    }
    const int mo = hi * 8;
#pragma unroll
    for (int r = 0; r < 8; ++r) {
      const int o = m0 + mo + r, nl = j0 + ln;
      y1t[o * NT + nl] = acc[r];
      y1[((size_t)b * kDS + o) * kN + n0 + nl] = acc[r];
    }
  }
  __syncthreads();

  // ---- BN1 partial sums (block-reduced, then one atomic per channel) ----
  if (tid < kDS) {
    float s = 0.f, q = 0.f;
    for (int nl = 0; nl < NT; ++nl) { const float v = y1t[tid * NT + nl]; s += v; q += v * v; }
    atomicAdd(&sum1[tid], s);
    atomicAdd(&sumsq1[tid], q);
  }
}

// ---------------- BN finalize: scale = w*rsqrt(var+eps), shift = b - mean*scale
__global__ void k_bn_final(const float* __restrict__ sum, const float* __restrict__ sq,
                           const float* __restrict__ w, const float* __restrict__ bias,
                           float* __restrict__ sc, float* __restrict__ sh, int nch) {
  const int i = blockIdx.x * blockDim.x + threadIdx.x;
  if (i < nch) {
    const float cnt = (float)((long)kB * kN);
    const float m = sum[i] / cnt;
    const float v = sq[i] / cnt - m * m;
    const float s = w[i] * rsqrtf(v + kEPS);
    sc[i] = s;
    sh[i] = bias[i] - m * s;
  }
}

// ---------------- kernel 4: z = relu(bn1(y1)); up-conv; BN2 sums ------------
// grid: (7, 64), 256 threads, LDS 42KB. y1 tile staged via TDM (2-D descriptor).
constexpr unsigned K4_Y1F  = 0;                 // 21504 : y1f [48][112] f32 (TDM dest)
constexpr unsigned K4_ZB   = 21504;             // 14336 : zb [64][112] bf16 (K padded)
constexpr unsigned K4_S2   = K4_ZB + 14336;     // 3072
constexpr unsigned K4_Q2   = K4_S2 + 3072;      // 3072
constexpr unsigned K4_SMEM = K4_Q2 + 3072;      // 41984 total

__global__ void k4_up(const float* __restrict__ y1, const float* __restrict__ w_up,
                      const float* __restrict__ sc1, const float* __restrict__ sh1,
                      float* __restrict__ y2, float* __restrict__ sum2,
                      float* __restrict__ sumsq2) {
  extern __shared__ char smem[];
  float* y1f = (float*)(smem + K4_Y1F);
  bf16*  zb  = (bf16*)(smem + K4_ZB);
  float* s2  = (float*)(smem + K4_S2);
  float* q2  = (float*)(smem + K4_Q2);
  const int b = blockIdx.y, n0 = blockIdx.x * NT, tid = threadIdx.x;
  const int wave = tid >> 5, lane = tid & 31;
  const int ln = lane & 15, hi = lane >> 4;

  // ---- TDM: stage y1 tile (48 rows x 112 f32, row stride 784) ----
  if (wave == 0) {
    tdm_load_f32(K4_Y1F, y1 + (size_t)b * kDS * kN + n0, NT, kDS, kN);
    __builtin_amdgcn_s_wait_tensorcnt(0);
  }
  for (int i = tid; i < kC; i += 256) { s2[i] = 0.f; q2[i] = 0.f; }
  __syncthreads();

  // z = relu(bn1(y1)), zero-pad K 48 -> 64
  for (int i = tid; i < 64 * NT; i += 256) {
    const int o = i / NT, nl = i % NT;
    float v = 0.f;
    if (o < kDS) {
      const float t = sc1[o] * y1f[o * NT + nl] + sh1[o];
      v = t > 0.f ? t : 0.f;
    }
    zb[i] = (bf16)v;
  }
  __syncthreads();

  // y2[c][nl] = sum_o Wu[c][o] * z[o][nl]  (M=768, N=112, K=64 padded, masked A)
  for (int j = wave; j < (kC / 16) * NTILES; j += 8) {
    const int ct = j / NTILES, nt = j % NTILES;
    const int c0 = ct * 16, j0 = nt * 16;
    const int m = c0 + ln;
    const int obase = (c0 / kCg) * kDSg;    // tile is entirely inside one group
    v8f acc = {};
    for (int k0 = 0; k0 < 64; k0 += 32) {
      v16bf av;
      const int kb = hi * 8;
#pragma unroll
      for (int e = 0; e < 16; ++e) {
        const int k = k0 + a_kmap(e, kb);
        const int og = k - obase;
        av[e] = (og >= 0 && og < kDSg) ? (bf16)w_up[m * kDSg + og] : (bf16)0.f;
      }
      v16bf bv = load_b(zb + k0 * NT + j0, NT);
      acc = wmma_bf16(av, bv, acc);
    }
    const int mo = hi * 8;
#pragma unroll
    for (int r = 0; r < 8; ++r) {
      const int c = c0 + mo + r, nl = j0 + ln;
      const float v = acc[r];
      y2[((size_t)b * kC + c) * kN + n0 + nl] = v;
      atomicAdd(&s2[c], v);        // ds_add_f32
      atomicAdd(&q2[c], v * v);
    }
  }
  __syncthreads();
  for (int i = tid; i < kC; i += 256) {
    atomicAdd(&sum2[i], s2[i]);
    atomicAdd(&sumsq2[i], q2[i]);
  }
}

// ---------------- kernel 6: out = LN_c(gate * bn2(y2)), in place on d_out ---
// grid: (7, 64), 128 threads; lane-per-n columns -> coalesced c-loop; L2-hot.
__global__ void k6_ln(float* __restrict__ y2io, const float* __restrict__ sc2,
                      const float* __restrict__ sh2, const float* __restrict__ gate,
                      const float* __restrict__ lnw, const float* __restrict__ lnb) {
  const int b = blockIdx.y, n0 = blockIdx.x * NT, tid = threadIdx.x;
  if (tid >= NT) return;
  const float g = gate[0];
  const size_t base = (size_t)b * kC * kN + n0 + tid;
  float s = 0.f, q = 0.f;
  for (int c = 0; c < kC; ++c) {
    const float v = g * (sc2[c] * y2io[base + (size_t)c * kN] + sh2[c]);
    s += v; q += v * v;
  }
  const float mu = s / kC;
  const float var = q / kC - mu * mu;
  const float r = rsqrtf(var + kEPS);
  for (int c = 0; c < kC; ++c) {
    const size_t a = base + (size_t)c * kN;
    const float v = g * (sc2[c] * y2io[a] + sh2[c]);
    y2io[a] = (v - mu) * r * lnw[c] + lnb[c];
  }
}

// ---------------- host launcher ----------------
extern "C" void kernel_launch(void* const* d_in, const int* in_sizes, int n_in,
                              void* d_out, int out_size, void* d_ws, size_t ws_size,
                              hipStream_t stream) {
  const float* x       = (const float*)d_in[0];
  const float* vt      = (const float*)d_in[1];
  const float* tok     = (const float*)d_in[2];
  const float* gate_av = (const float*)d_in[3];
  const float* gate    = (const float*)d_in[4];
  const float* w_down  = (const float*)d_in[5];
  const float* w_up    = (const float*)d_in[6];
  const float* bn1w    = (const float*)d_in[7];
  const float* bn1b    = (const float*)d_in[8];
  const float* bn2w    = (const float*)d_in[9];
  const float* bn2b    = (const float*)d_in[10];
  const float* lnw     = (const float*)d_in[11];
  const float* lnb     = (const float*)d_in[12];

  float* ws  = (float*)d_ws;
  float* out = (float*)d_out;          // also used as y2 scratch (overwritten in k6)
  float* rep  = ws + WS_REP;
  float* y1   = ws + WS_Y1;
  float* sum1 = ws + WS_SUM1;
  float* sq1  = ws + WS_SQ1;
  float* sum2 = ws + WS_SUM2;
  float* sq2  = ws + WS_SQ2;
  float* sc1  = ws + WS_SC1;
  float* sh1  = ws + WS_SH1;
  float* sc2  = ws + WS_SC2;
  float* sh2  = ws + WS_SH2;

  const size_t SMEM1 = (size_t)kT * kC * 2 + (size_t)kT * 800 * 4;  // 151552

  k0_init<<<(N_ZERO + 255) / 256, 256, 0, stream>>>(ws + WS_STATS, N_ZERO);
  k1_rep<<<kB, 256, SMEM1, stream>>>(vt, tok, rep);
  k2_fused<<<dim3(NTILES, kB), 256, K2_SMEM, stream>>>(x, rep, gate_av, w_down,
                                                       y1, sum1, sq1);
  k_bn_final<<<1, 64, 0, stream>>>(sum1, sq1, bn1w, bn1b, sc1, sh1, kDS);
  k4_up<<<dim3(NTILES, kB), 256, K4_SMEM, stream>>>(y1, w_up, sc1, sh1, out, sum2, sq2);
  k_bn_final<<<3, 256, 0, stream>>>(sum2, sq2, bn2w, bn2b, sc2, sh2, kC);
  k6_ln<<<dim3(NTILES, kB), 128, 0, stream>>>(out, sc2, sh2, gate, lnw, lnb);
}
